// MPNNLayer_72808285602199
// MI455X (gfx1250) — matrix operand order repaired and checked by reading, not compiled, not run on hardware
//
#include <hip/hip_runtime.h>
#include <hip/hip_bf16.h>

typedef __attribute__((ext_vector_type(16))) __bf16 v16bf;
typedef __attribute__((ext_vector_type(8)))  __bf16 v8bf;
typedef __attribute__((ext_vector_type(8)))  float  v8f;

#define D_DIM 128
#define EPSV 1e-5f
#define N_NODES 20000
#define N_EDGES 640000

// ---- workspace layout (bytes) ----
static constexpr size_t OFF_MSG  = 0;                               // [20000*128] f32
static constexpr size_t OFF_CNT  = OFF_MSG + (size_t)N_NODES * D_DIM * 4; // [20000] f32
static constexpr size_t OFF_W1T  = OFF_CNT + (size_t)N_NODES * 4;   // [128][384] bf16
static constexpr size_t OFF_W2T  = OFF_W1T + 49152 * 2;             // [128][128] bf16 (contiguous w/ W1T)
static constexpr size_t OFF_W3T  = OFF_W2T + 16384 * 2;             // [128][128] bf16
static constexpr size_t OFF_WD1T = OFF_W3T + 16384 * 2;             // [512][128] bf16
static constexpr size_t OFF_WD2T = OFF_WD1T + 65536 * 2;            // [128][512] bf16

// ---------------- helpers ----------------
// Load 16 contiguous bf16 (B-fragment: lane holds col n, K = klo..klo+15 contiguous).
__device__ __forceinline__ v16bf ld16(const __bf16* p) {
  v8bf lo = *(const v8bf*)p;
  v8bf hh = *(const v8bf*)(p + 8);
  v16bf r;
#pragma unroll
  for (int i = 0; i < 8; ++i) { r[i] = lo[i]; r[i + 8] = hh[i]; }
  return r;
}

// A-fragment from a row-major [16][stride] bf16 tile in LDS.
// Contiguous layout: elems 0..15 -> K = kb+0..15, kb = hi*16.
__device__ __forceinline__ v16bf ld_afrag(const __bf16* st, int row, int kt, int hi, int stride) {
  const __bf16* p = st + row * stride + kt * 32 + hi * 16;
  v8bf lo = *(const v8bf*)p;
  v8bf hh = *(const v8bf*)(p + 8);
  v16bf r;
#pragma unroll
  for (int i = 0; i < 8; ++i) { r[i] = lo[i]; r[i + 8] = hh[i]; }
  return r;
}

__device__ __forceinline__ v8f wmma_bf16(v16bf a, v16bf b, v8f c) {
  return __builtin_amdgcn_wmma_f32_16x16x32_bf16(false, a, false, b, (short)0, c, false, false);
}

// ---------------- kernel 1: weight convert + transpose to bf16 [N][K] ----------------
__global__ void prep_weights(const float* __restrict__ W1, const float* __restrict__ W2,
                             const float* __restrict__ W3, const float* __restrict__ Wd1,
                             const float* __restrict__ Wd2,
                             __bf16* __restrict__ w1t, __bf16* __restrict__ w2t,
                             __bf16* __restrict__ w3t, __bf16* __restrict__ wd1t,
                             __bf16* __restrict__ wd2t) {
  int i = blockIdx.x * blockDim.x + threadIdx.x;
  if (i < 49152) {                       // W1 [384][128] -> w1t [128][384]
    int n = i / 384, k = i % 384;
    w1t[i] = (__bf16)W1[k * 128 + n];
  } else if (i < 65536) {                // W2 [128][128] -> w2t [128][128]
    int j = i - 49152; int n = j / 128, k = j % 128;
    w2t[j] = (__bf16)W2[k * 128 + n];
  } else if (i < 81920) {                // W3
    int j = i - 65536; int n = j / 128, k = j % 128;
    w3t[j] = (__bf16)W3[k * 128 + n];
  } else if (i < 147456) {               // Wd1 [128][512] -> wd1t [512][128]
    int j = i - 81920; int n = j / 128, k = j % 128;
    wd1t[j] = (__bf16)Wd1[k * 512 + n];
  } else if (i < 212992) {               // Wd2 [512][128] -> wd2t [128][512]
    int j = i - 147456; int n = j / 512, k = j % 512;
    wd2t[j] = (__bf16)Wd2[k * 128 + n];
  }
}

// ---------------- kernel 2: edge message MLP + atomic scatter ----------------
// LDS: W1t/W2t/W3t (81920 bf16 = 163840B) + per-wave stageA/stageB (2x 16x128 bf16 = 8KB/wave)
#define EW_WAVES 8
#define EDGE_LDS (163840 + EW_WAVES * 8192)

__global__ __launch_bounds__(EW_WAVES * 32) void
edge_kernel(const float* __restrict__ hV, const float* __restrict__ hE,
            const int* __restrict__ eidx, const float* __restrict__ b1,
            const float* __restrict__ b2, const float* __restrict__ b3,
            const __bf16* __restrict__ w1t,  // W1t..W3t contiguous in ws
            float* __restrict__ msg, float* __restrict__ cnt) {
  extern __shared__ char smem[];
  __bf16* sW1 = (__bf16*)smem;           // [128][384]
  __bf16* sW2 = sW1 + 49152;             // [128][128]
  __bf16* sW3 = sW2 + 16384;             // [128][128]
  int w = threadIdx.x >> 5;
  __bf16* stA = (__bf16*)(smem + 163840 + w * 8192);      // [16][128]
  __bf16* stB = stA + 2048;                                // [16][128]

  // cooperative load of all 3 weight matrices (contiguous, 163840 B)
  {
    const uint4* src = (const uint4*)w1t;
    uint4* dst = (uint4*)smem;
    for (int i = threadIdx.x; i < 163840 / 16; i += blockDim.x) dst[i] = src[i];
  }
  __syncthreads();

  const int lane = threadIdx.x & 31;
  const int m = lane & 15;       // row within 16-edge tile / col within N-tile
  const int hi = lane >> 4;      // K-half selector
  const int wave = blockIdx.x * EW_WAVES + w;
  const int nwaves = gridDim.x * EW_WAVES;

  for (int tile = wave; tile < N_EDGES / 16; tile += nwaves) {
    int e = tile * 16 + m;
    int src = eidx[e];
    int tgt = eidx[N_EDGES + e];
    const float* rowE = hE + (size_t)e * D_DIM;
    const float* rowS = hV + (size_t)src * D_DIM;
    const float* rowT = hV + (size_t)tgt * D_DIM;

    // Build A fragments for the [16 x 384] gathered tile, bf16, contiguous K layout (k = hi*16 + s).
    v16bf A[12];
#pragma unroll
    for (int kt = 0; kt < 12; ++kt) {
      const float* base = (kt < 4) ? rowE : (kt < 8 ? rowS : rowT);
      const float* p = base + ((kt & 3) * 32 + hi * 16);
      float4 a0 = *(const float4*)(p);
      float4 a1 = *(const float4*)(p + 4);
      float4 c0 = *(const float4*)(p + 8);
      float4 c1 = *(const float4*)(p + 12);
      v16bf a;
      a[0] = (__bf16)a0.x; a[1] = (__bf16)a0.y; a[2] = (__bf16)a0.z; a[3] = (__bf16)a0.w;
      a[4] = (__bf16)a1.x; a[5] = (__bf16)a1.y; a[6] = (__bf16)a1.z; a[7] = (__bf16)a1.w;
      a[8] = (__bf16)c0.x; a[9] = (__bf16)c0.y; a[10] = (__bf16)c0.z; a[11] = (__bf16)c0.w;
      a[12] = (__bf16)c1.x; a[13] = (__bf16)c1.y; a[14] = (__bf16)c1.z; a[15] = (__bf16)c1.w;
      A[kt] = a;
    }

    // GEMM1: [16x384] @ [384x128], relu, restage bf16 (A-layout for next GEMM)
#pragma unroll
    for (int nt = 0; nt < 8; ++nt) {
      v8f acc = {0.f, 0.f, 0.f, 0.f, 0.f, 0.f, 0.f, 0.f};
      int n = nt * 16 + m;
      const __bf16* bcol = sW1 + n * 384 + hi * 16;
#pragma unroll
      for (int kt = 0; kt < 12; ++kt)
        acc = wmma_bf16(A[kt], ld16(bcol + kt * 32), acc);
      float bias = b1[n];
#pragma unroll
      for (int r = 0; r < 8; ++r)
        stA[(r + hi * 8) * 128 + n] = (__bf16)fmaxf(acc[r] + bias, 0.f);
    }

    // GEMM2: [16x128] @ [128x128], relu
    v16bf A2[4];
#pragma unroll
    for (int kt = 0; kt < 4; ++kt) A2[kt] = ld_afrag(stA, m, kt, hi, 128);
#pragma unroll
    for (int nt = 0; nt < 8; ++nt) {
      v8f acc = {0.f, 0.f, 0.f, 0.f, 0.f, 0.f, 0.f, 0.f};
      int n = nt * 16 + m;
      const __bf16* bcol = sW2 + n * 128 + hi * 16;
#pragma unroll
      for (int kt = 0; kt < 4; ++kt)
        acc = wmma_bf16(A2[kt], ld16(bcol + kt * 32), acc);
      float bias = b2[n];
#pragma unroll
      for (int r = 0; r < 8; ++r)
        stB[(r + hi * 8) * 128 + n] = (__bf16)fmaxf(acc[r] + bias, 0.f);
    }

    // GEMM3: message = [16x128] @ [128x128] + b3, atomic scatter to msg_sum[tgt]
    v16bf A3[4];
#pragma unroll
    for (int kt = 0; kt < 4; ++kt) A3[kt] = ld_afrag(stB, m, kt, hi, 128);
#pragma unroll
    for (int nt = 0; nt < 8; ++nt) {
      v8f acc = {0.f, 0.f, 0.f, 0.f, 0.f, 0.f, 0.f, 0.f};
      int n = nt * 16 + m;
      const __bf16* bcol = sW3 + n * 128 + hi * 16;
#pragma unroll
      for (int kt = 0; kt < 4; ++kt)
        acc = wmma_bf16(A3[kt], ld16(bcol + kt * 32), acc);
      float bias = b3[n];
#pragma unroll
      for (int r = 0; r < 8; ++r) {
        int row = r + hi * 8;                        // D-layout row for this VGPR
        int t = __shfl(tgt, row, 32);                // tgt of edge 'row' (held by lane 'row')
        unsafeAtomicAdd(msg + (size_t)t * D_DIM + n, acc[r] + bias);
      }
    }
    if (hi == 0) unsafeAtomicAdd(cnt + tgt, 1.0f);   // one count per edge
  }
}

// ---------------- kernel 3: node update (mean, LN1, FFN, LN2) ----------------
#define NW_WAVES 4
#define NODE_WAVE_LDS 36864  // ln bf16 4KB + resid f32 8KB + h1 bf16 16KB + out f32 8KB
#define NODE_LDS (NW_WAVES * NODE_WAVE_LDS)

__global__ __launch_bounds__(NW_WAVES * 32) void
node_kernel(const float* __restrict__ hV, const float* __restrict__ msg,
            const float* __restrict__ cnt, const float* __restrict__ bd1,
            const float* __restrict__ bd2, const float* __restrict__ g1,
            const float* __restrict__ be1, const float* __restrict__ g2,
            const float* __restrict__ be2, const __bf16* __restrict__ wd1t,
            const __bf16* __restrict__ wd2t, float* __restrict__ out) {
  extern __shared__ char smem[];
  int w = threadIdx.x >> 5;
  char* base = smem + w * NODE_WAVE_LDS;
  __bf16* lnS   = (__bf16*)base;            // [16][128] bf16
  float*  resid = (float*)(base + 4096);    // [16][128] f32
  __bf16* h1S   = (__bf16*)(base + 12288);  // [16][512] bf16
  float*  outS  = (float*)(base + 28672);   // [16][128] f32

  const int lane = threadIdx.x & 31;
  const int m = lane & 15;
  const int hi = lane >> 4;
  const int wave = blockIdx.x * NW_WAVES + w;
  const int nwaves = gridDim.x * NW_WAVES;

  for (int tile = wave; tile < N_NODES / 16; tile += nwaves) {
    // ---- dh = msg/cnt ; x = hV + dh ; LN1 -> lnS(bf16) + resid(f32) ----
#pragma unroll 1
    for (int r = 0; r < 16; ++r) {
      int node = tile * 16 + r;
      int c0 = lane * 4;
      float4 hv = *(const float4*)(hV + (size_t)node * D_DIM + c0);
      float4 ms = *(const float4*)(msg + (size_t)node * D_DIM + c0);
      float invc = 1.0f / cnt[node];
      float x0 = hv.x + ms.x * invc, x1 = hv.y + ms.y * invc;
      float x2 = hv.z + ms.z * invc, x3 = hv.w + ms.w * invc;
      float s = x0 + x1 + x2 + x3;
      float q = x0 * x0 + x1 * x1 + x2 * x2 + x3 * x3;
#pragma unroll
      for (int o = 16; o >= 1; o >>= 1) {
        s += __shfl_xor(s, o, 32);
        q += __shfl_xor(q, o, 32);
      }
      float mu = s * (1.f / 128.f);
      float var = q * (1.f / 128.f) - mu * mu;
      float rs = rsqrtf(var + EPSV);
      float4 gg = *(const float4*)(g1 + c0);
      float4 bb = *(const float4*)(be1 + c0);
      float y0 = (x0 - mu) * rs * gg.x + bb.x;
      float y1 = (x1 - mu) * rs * gg.y + bb.y;
      float y2 = (x2 - mu) * rs * gg.z + bb.z;
      float y3 = (x3 - mu) * rs * gg.w + bb.w;
      lnS[r * 128 + c0 + 0] = (__bf16)y0; lnS[r * 128 + c0 + 1] = (__bf16)y1;
      lnS[r * 128 + c0 + 2] = (__bf16)y2; lnS[r * 128 + c0 + 3] = (__bf16)y3;
      *(float4*)(resid + r * 128 + c0) = float4{y0, y1, y2, y3};
    }

    // ---- FFN1: [16x128] @ [128x512] + bd1, relu -> h1S (B streamed from L2) ----
    v16bf A1[4];
#pragma unroll
    for (int kt = 0; kt < 4; ++kt) A1[kt] = ld_afrag(lnS, m, kt, hi, 128);
#pragma unroll
    for (int nt = 0; nt < 32; ++nt) {
      v8f acc = {0.f, 0.f, 0.f, 0.f, 0.f, 0.f, 0.f, 0.f};
      int n = nt * 16 + m;
      const __bf16* bcol = wd1t + n * 128 + hi * 16;
#pragma unroll
      for (int kt = 0; kt < 4; ++kt)
        acc = wmma_bf16(A1[kt], ld16(bcol + kt * 32), acc);
      float bias = bd1[n];
#pragma unroll
      for (int r = 0; r < 8; ++r)
        h1S[(r + hi * 8) * 512 + n] = (__bf16)fmaxf(acc[r] + bias, 0.f);
    }

    // ---- FFN2: [16x512] @ [512x128] + bd2 + resid -> outS ----
    v16bf A2[16];
#pragma unroll
    for (int kt = 0; kt < 16; ++kt) A2[kt] = ld_afrag(h1S, m, kt, hi, 512);
#pragma unroll
    for (int nt = 0; nt < 8; ++nt) {
      v8f acc = {0.f, 0.f, 0.f, 0.f, 0.f, 0.f, 0.f, 0.f};
      int n = nt * 16 + m;
      const __bf16* bcol = wd2t + n * 512 + hi * 16;
#pragma unroll
      for (int kt = 0; kt < 16; ++kt)
        acc = wmma_bf16(A2[kt], ld16(bcol + kt * 32), acc);
      float bias = bd2[n];
#pragma unroll
      for (int r = 0; r < 8; ++r) {
        int row = r + hi * 8;
        outS[row * 128 + n] = acc[r] + bias + resid[row * 128 + n];
      }
    }

    // ---- LN2 -> out ----
#pragma unroll 1
    for (int r = 0; r < 16; ++r) {
      int node = tile * 16 + r;
      int c0 = lane * 4;
      float4 x = *(const float4*)(outS + r * 128 + c0);
      float s = x.x + x.y + x.z + x.w;
      float q = x.x * x.x + x.y * x.y + x.z * x.z + x.w * x.w;
#pragma unroll
      for (int o = 16; o >= 1; o >>= 1) {
        s += __shfl_xor(s, o, 32);
        q += __shfl_xor(q, o, 32);
      }
      float mu = s * (1.f / 128.f);
      float var = q * (1.f / 128.f) - mu * mu;
      float rs = rsqrtf(var + EPSV);
      float4 gg = *(const float4*)(g2 + c0);
      float4 bb = *(const float4*)(be2 + c0);
      float4 y;
      y.x = (x.x - mu) * rs * gg.x + bb.x;
      y.y = (x.y - mu) * rs * gg.y + bb.y;
      y.z = (x.z - mu) * rs * gg.z + bb.z;
      y.w = (x.w - mu) * rs * gg.w + bb.w;
      *(float4*)(out + (size_t)node * D_DIM + c0) = y;
    }
  }
}

// ---------------- launch ----------------
extern "C" void kernel_launch(void* const* d_in, const int* in_sizes, int n_in,
                              void* d_out, int out_size, void* d_ws, size_t ws_size,
                              hipStream_t stream) {
  (void)in_sizes; (void)n_in; (void)out_size; (void)ws_size;
  const float* hV  = (const float*)d_in[0];
  const float* hE  = (const float*)d_in[1];
  const int*  eidx = (const int*)d_in[2];
  const float* W1  = (const float*)d_in[3];
  const float* b1  = (const float*)d_in[4];
  const float* W2  = (const float*)d_in[5];
  const float* b2  = (const float*)d_in[6];
  const float* W3  = (const float*)d_in[7];
  const float* b3  = (const float*)d_in[8];
  const float* Wd1 = (const float*)d_in[9];
  const float* bd1 = (const float*)d_in[10];
  const float* Wd2 = (const float*)d_in[11];
  const float* bd2 = (const float*)d_in[12];
  const float* g1  = (const float*)d_in[13];
  const float* be1 = (const float*)d_in[14];
  const float* g2  = (const float*)d_in[15];
  const float* be2 = (const float*)d_in[16];

  char* ws = (char*)d_ws;
  float*  msg  = (float*)(ws + OFF_MSG);
  float*  cnt  = (float*)(ws + OFF_CNT);
  __bf16* w1t  = (__bf16*)(ws + OFF_W1T);
  __bf16* w2t  = (__bf16*)(ws + OFF_W2T);
  __bf16* w3t  = (__bf16*)(ws + OFF_W3T);
  __bf16* wd1t = (__bf16*)(ws + OFF_WD1T);
  __bf16* wd2t = (__bf16*)(ws + OFF_WD2T);

  // zero msg_sum + cnt (contiguous at start of ws)
  hipMemsetAsync(ws, 0, OFF_CNT + (size_t)N_NODES * 4, stream);

  prep_weights<<<(212992 + 255) / 256, 256, 0, stream>>>(W1, W2, W3, Wd1, Wd2,
                                                         w1t, w2t, w3t, wd1t, wd2t);

  edge_kernel<<<512, EW_WAVES * 32, EDGE_LDS, stream>>>(hV, hE, eidx, b1, b2, b3,
                                                        w1t, msg, cnt);

  node_kernel<<<320, NW_WAVES * 32, NODE_LDS, stream>>>(hV, msg, cnt, bd1, bd2,
                                                        g1, be1, g2, be2,
                                                        wd1t, wd2t, (float*)d_out);
}
